// LigerFusedLinearJSDLoss_71382356459699
// MI455X (gfx1250) — compile-verified, matrix operand order checked by reading, not run on hardware
//
#include <hip/hip_runtime.h>
#include <hip/hip_bf16.h>
#include <math.h>

// ---------------- problem constants ----------------
#define N_TOK 2048
#define H_STU 2048
#define H_TEA 4096
#define VOCAB 32000

// ---------------- tiling ----------------
#define NBV       16                 // vocab columns per WMMA n-tile
#define CHUNK_NB  16                 // n-tiles per workgroup (vocab chunk = 256)
#define CHUNK_V   (NBV * CHUNK_NB)   // 256
#define NCHUNK    (VOCAB / CHUNK_V)  // 125 workgroups
#define MTILES    (N_TOK / 16)       // 128 token tiles
#define WAVES     8
#define TPB       (WAVES * 32)       // 256 threads = 8 wave32
#define NEG_HUGE  (-3.0e38f)

#define LDS_PAD   8                  // bf16 elements of row padding (16B) -> kills
#define PITCH_S   (H_STU + LDS_PAD)  // 16-way LDS bank conflicts on B-frag reads
#define PITCH_T   (H_TEA + LDS_PAD)

typedef __bf16 bf16;
typedef __attribute__((ext_vector_type(16))) __bf16 v16bf;
typedef __attribute__((ext_vector_type(8)))  float  v8f;

// ---------------- helpers ----------------
__device__ __forceinline__ bf16 f2bf(float x) {
  // round-to-nearest-even fp32 -> bf16
  unsigned u = __float_as_uint(x);
  unsigned r = (u + 0x7FFFu + ((u >> 16) & 1u)) >> 16;
  unsigned short s = (unsigned short)r;
  return __builtin_bit_cast(bf16, s);
}

// Load one 16x32 bf16 WMMA fragment (A-layout; B uses the same pattern with
// row = vocab column since W is row-major over K = B^T column-major).
// lane 0-15 : rows, K in {kb..kb+7} ++ {kb+16..kb+23}
// lane 16-31: rows, K in {kb+8..kb+15} ++ {kb+24..kb+31}
__device__ __forceinline__ v16bf load_frag(const bf16* __restrict__ base, int ld,
                                           int row, int kb, int lane) {
  int khalf = (lane >> 4) << 3;
  const bf16* p = base + (size_t)row * ld + kb + khalf;
  union { uint4 u; bf16 h[8]; } lo, hi;
  lo.u = *(const uint4*)(p);
  hi.u = *(const uint4*)(p + 16);
  v16bf f;
#pragma unroll
  for (int j = 0; j < 8; ++j) { f[j] = lo.h[j]; f[8 + j] = hi.h[j]; }
  return f;
}

// Stage NBV weight rows (fp32, row-major over H) into LDS as bf16 with padded pitch.
__device__ __forceinline__ void stage_w(bf16* __restrict__ dst,
                                        const float* __restrict__ src,
                                        int v0, int H, int pitch, int tid) {
  const int total = NBV * H;
  for (int e = tid * 4; e < total; e += TPB * 4) {
    int rr = e / H, cc = e - rr * H;
    float4 f = *(const float4*)(src + (size_t)(v0 + rr) * H + cc);
    bf16* d = dst + (size_t)rr * pitch + cc;
    d[0] = f2bf(f.x); d[1] = f2bf(f.y); d[2] = f2bf(f.z); d[3] = f2bf(f.w);
  }
}

// Two 16x16 logit tiles sharing one B fragment; software-pipelined K-loop so
// next-chunk loads overlap the two independent v_wmma streams.
__device__ __forceinline__ void gemm_tile2(const bf16* __restrict__ act,
                                           const bf16* __restrict__ wlds,
                                           int H, int ldw, int r0, int r1,
                                           int lane, v8f& c0, v8f& c1) {
  const int col = lane & 15;
  v16bf a0 = load_frag(act,  H,   r0,  0, lane);
  v16bf a1 = load_frag(act,  H,   r1,  0, lane);
  v16bf b  = load_frag(wlds, ldw, col, 0, lane);
#pragma unroll 2
  for (int kb = 32; kb < H; kb += 32) {
    v16bf na0 = load_frag(act,  H,   r0,  kb, lane);
    v16bf na1 = load_frag(act,  H,   r1,  kb, lane);
    v16bf nb  = load_frag(wlds, ldw, col, kb, lane);
    c0 = __builtin_amdgcn_wmma_f32_16x16x32_bf16(false, a0, false, b, (short)0, c0, false, false);
    c1 = __builtin_amdgcn_wmma_f32_16x16x32_bf16(false, a1, false, b, (short)0, c1, false, false);
    a0 = na0; a1 = na1; b = nb;
  }
  c0 = __builtin_amdgcn_wmma_f32_16x16x32_bf16(false, a0, false, b, (short)0, c0, false, false);
  c1 = __builtin_amdgcn_wmma_f32_16x16x32_bf16(false, a1, false, b, (short)0, c1, false, false);
}

// Online (max, sumexp) update into LDS stats from a C fragment.
// C layout: element c[i] of lane l holds M = i + 8*(l>=16), N = l&15.
__device__ __forceinline__ void tile_update(v8f c, float* st, int m0, int lane, int offs) {
#pragma unroll
  for (int i = 0; i < 8; ++i) {
    float v = c[i];
    float mx = v;
#pragma unroll
    for (int d = 8; d; d >>= 1) mx = fmaxf(mx, __shfl_xor(mx, d, 32));
    float e = __expf(v - mx);
#pragma unroll
    for (int d = 8; d; d >>= 1) e += __shfl_xor(e, d, 32);
    if ((lane & 15) == i) {           // lane i -> row m0+i ; lane 16+i -> row m0+8+i
      int tok = m0 + i + ((lane >> 4) << 3);
      float om = st[4 * tok + offs], oz = st[4 * tok + offs + 1];
      float nm = fmaxf(om, mx);
      float nz = oz * __expf(om - nm) + e * __expf(mx - nm);
      st[4 * tok + offs]     = nm;
      st[4 * tok + offs + 1] = nz;
    }
  }
}

// ---------------- kernels ----------------
__global__ void cvt_kernel(const float* __restrict__ in, bf16* __restrict__ out, int n4) {
  for (int i = blockIdx.x * blockDim.x + threadIdx.x; i < n4; i += gridDim.x * blockDim.x) {
    float4 f = *(const float4*)(in + (size_t)i * 4);
    bf16* o = out + (size_t)i * 4;
    o[0] = f2bf(f.x); o[1] = f2bf(f.y); o[2] = f2bf(f.z); o[3] = f2bf(f.w);
  }
}

__global__ __launch_bounds__(TPB) void pass1_kernel(
    const bf16* __restrict__ sact, const bf16* __restrict__ tact,
    const float* __restrict__ sw,  const float* __restrict__ tw,
    float* __restrict__ pm_s, float* __restrict__ pz_s,
    float* __restrict__ pm_t, float* __restrict__ pz_t) {
  extern __shared__ char smem[];
  bf16*  w_s = (bf16*)smem;
  bf16*  w_t = (bf16*)(smem + (size_t)NBV * PITCH_S * sizeof(bf16));
  float* st  = (float*)(smem + (size_t)NBV * (PITCH_S + PITCH_T) * sizeof(bf16));
  const int chunk = blockIdx.x;
  const int tid = threadIdx.x, lane = tid & 31, wave = tid >> 5;

  for (int i = tid; i < N_TOK; i += TPB) {
    st[4*i+0] = NEG_HUGE; st[4*i+1] = 0.f;
    st[4*i+2] = NEG_HUGE; st[4*i+3] = 0.f;
  }
  for (int nb = 0; nb < CHUNK_NB; ++nb) {
    int v0 = chunk * CHUNK_V + nb * NBV;
    __syncthreads();                       // prior tiles done (also covers init)
    stage_w(w_s, sw, v0, H_STU, PITCH_S, tid);
    stage_w(w_t, tw, v0, H_TEA, PITCH_T, tid);
    if (nb + 1 < CHUNK_NB) {               // hint next weight tile into cache
      __builtin_prefetch(sw + (size_t)(v0 + NBV) * H_STU + (tid << 4), 0, 1);
      __builtin_prefetch(tw + (size_t)(v0 + NBV) * H_TEA + (tid << 4), 0, 1);
    }
    __syncthreads();
    for (int mt2 = wave; mt2 < MTILES / 2; mt2 += WAVES) {
      int m0 = (mt2 * 2) << 4;
      int r0 = m0 + (lane & 15);
      int r1 = r0 + 16;
      v8f c0 = {}, c1 = {};
      gemm_tile2(sact, w_s, H_STU, PITCH_S, r0, r1, lane, c0, c1);
      tile_update(c0, st, m0,      lane, 0);
      tile_update(c1, st, m0 + 16, lane, 0);
      v8f d0 = {}, d1 = {};
      gemm_tile2(tact, w_t, H_TEA, PITCH_T, r0, r1, lane, d0, d1);
      tile_update(d0, st, m0,      lane, 2);
      tile_update(d1, st, m0 + 16, lane, 2);
    }
  }
  __syncthreads();
  for (int i = tid; i < N_TOK; i += TPB) {
    size_t o = (size_t)chunk * N_TOK + i;
    pm_s[o] = st[4*i+0]; pz_s[o] = st[4*i+1];
    pm_t[o] = st[4*i+2]; pz_t[o] = st[4*i+3];
  }
}

__global__ void lse_kernel(const float* __restrict__ pm, const float* __restrict__ pz,
                           float* __restrict__ lse) {
  int i = blockIdx.x * blockDim.x + threadIdx.x;
  if (i >= N_TOK) return;
  float M = NEG_HUGE, Z = 0.f;
  for (int c = 0; c < NCHUNK; ++c) {
    float m = pm[(size_t)c * N_TOK + i], z = pz[(size_t)c * N_TOK + i];
    float nm = fmaxf(M, m);
    Z = Z * __expf(M - nm) + z * __expf(m - nm);
    M = nm;
  }
  lse[i] = M + __logf(Z);
}

__global__ __launch_bounds__(TPB) void pass2_kernel(
    const bf16* __restrict__ sact, const bf16* __restrict__ tact,
    const float* __restrict__ sw,  const float* __restrict__ tw,
    const float* __restrict__ lse_s, const float* __restrict__ lse_t,
    const int* __restrict__ labels,
    float* __restrict__ part_j, float* __restrict__ part_h) {
  extern __shared__ char smem[];
  bf16*  w_s   = (bf16*)smem;
  bf16*  w_t   = (bf16*)(smem + (size_t)NBV * PITCH_S * sizeof(bf16));
  char*  tail  = smem + (size_t)NBV * (PITCH_S + PITCH_T) * sizeof(bf16);
  float* lse_sl = (float*)tail;
  float* lse_tl = lse_sl + N_TOK;
  int*   lbl_l  = (int*)(lse_tl + N_TOK);
  float* red    = (float*)(lbl_l + N_TOK);
  const int chunk = blockIdx.x;
  const int tid = threadIdx.x, lane = tid & 31, wave = tid >> 5;
  float accj = 0.f, acch = 0.f;

  for (int i = tid; i < N_TOK; i += TPB) {   // stage row stats + labels once
    lse_sl[i] = lse_s[i]; lse_tl[i] = lse_t[i]; lbl_l[i] = labels[i];
  }
  for (int nb = 0; nb < CHUNK_NB; ++nb) {
    int v0 = chunk * CHUNK_V + nb * NBV;
    __syncthreads();
    stage_w(w_s, sw, v0, H_STU, PITCH_S, tid);
    stage_w(w_t, tw, v0, H_TEA, PITCH_T, tid);
    __syncthreads();
    for (int mt2 = wave; mt2 < MTILES / 2; mt2 += WAVES) {
      int m0 = (mt2 * 2) << 4;
      int r0 = m0 + (lane & 15);
      int r1 = r0 + 16;
      v8f cs0 = {}, cs1 = {};
      gemm_tile2(sact, w_s, H_STU, PITCH_S, r0, r1, lane, cs0, cs1);
      v8f ct0 = {}, ct1 = {};
      gemm_tile2(tact, w_t, H_TEA, PITCH_T, r0, r1, lane, ct0, ct1);
      int vidx = v0 + (lane & 15);
#pragma unroll
      for (int t = 0; t < 2; ++t) {
        v8f cs = t ? cs1 : cs0;
        v8f ct = t ? ct1 : ct0;
        int mb = m0 + 16 * t;
#pragma unroll
        for (int i = 0; i < 8; ++i) {
          int tok = mb + i + ((lane >> 4) << 3);
          float slp = cs[i] - lse_sl[tok];
          float tlp = ct[i] - lse_tl[tok];
          float p = __expf(slp), q = __expf(tlp);
          float lm = __logf(0.5f * (p + q));
          accj += 0.5f * (q * (tlp - lm) + p * (slp - lm));   // BETA = 0.5 JSD term
          if (lbl_l[tok] == vidx) acch -= slp;                // NLL gather
        }
      }
    }
  }
#pragma unroll
  for (int d = 16; d; d >>= 1) {
    accj += __shfl_xor(accj, d, 32);
    acch += __shfl_xor(acch, d, 32);
  }
  if (lane == 0) { red[2 * wave] = accj; red[2 * wave + 1] = acch; }
  __syncthreads();
  if (tid == 0) {
    float j = 0.f, h = 0.f;
    for (int w2 = 0; w2 < WAVES; ++w2) { j += red[2 * w2]; h += red[2 * w2 + 1]; }
    part_j[chunk] = j; part_h[chunk] = h;
  }
}

__global__ void finalize_kernel(const float* __restrict__ part_j,
                                const float* __restrict__ part_h,
                                float* __restrict__ out) {
  if (threadIdx.x == 0 && blockIdx.x == 0) {
    float j = 0.f, h = 0.f;
    for (int c = 0; c < NCHUNK; ++c) { j += part_j[c]; h += part_h[c]; }
    float inv = 1.0f / (float)N_TOK;
    out[0] = 0.5f * (h * inv) + 0.5f * (j * inv);
  }
}

// ---------------- host launcher ----------------
extern "C" void kernel_launch(void* const* d_in, const int* in_sizes, int n_in,
                              void* d_out, int out_size, void* d_ws, size_t ws_size,
                              hipStream_t stream) {
  (void)in_sizes; (void)n_in; (void)out_size; (void)ws_size;
  const float* s_in  = (const float*)d_in[0];
  const float* s_w   = (const float*)d_in[1];
  const float* t_in  = (const float*)d_in[2];
  const float* t_w   = (const float*)d_in[3];
  const int*   lbl   = (const int*)d_in[4];
  float* out = (float*)d_out;

  char* ws = (char*)d_ws;
  size_t o = 0;
  auto take = [&](size_t bytes) {
    o = (o + 255) & ~(size_t)255;
    char* p = ws + o; o += bytes; return p;
  };
  bf16*  sact  = (bf16*)take((size_t)N_TOK * H_STU * sizeof(bf16)); //  8 MB
  bf16*  tact  = (bf16*)take((size_t)N_TOK * H_TEA * sizeof(bf16)); // 16 MB
  float* pm_s  = (float*)take((size_t)NCHUNK * N_TOK * sizeof(float));
  float* pz_s  = (float*)take((size_t)NCHUNK * N_TOK * sizeof(float));
  float* pm_t  = (float*)take((size_t)NCHUNK * N_TOK * sizeof(float));
  float* pz_t  = (float*)take((size_t)NCHUNK * N_TOK * sizeof(float));
  float* lse_s = (float*)take((size_t)N_TOK * sizeof(float));
  float* lse_t = (float*)take((size_t)N_TOK * sizeof(float));
  float* prt_j = (float*)take((size_t)NCHUNK * sizeof(float));
  float* prt_h = (float*)take((size_t)NCHUNK * sizeof(float));

  size_t wbytes = (size_t)NBV * (PITCH_S + PITCH_T) * sizeof(bf16);     // 197120
  size_t sh1 = wbytes + (size_t)N_TOK * 4 * sizeof(float);              // +32KB stats
  size_t sh2 = wbytes + (size_t)N_TOK * 3 * sizeof(float)               // lse_s/t + labels
             + 2 * WAVES * sizeof(float);
  // Allow >64KB dynamic LDS (CDNA5 WGP has 320KB); ignore errors on toolchains
  // where the cap is already the hardware limit.
  (void)hipFuncSetAttribute((const void*)pass1_kernel,
                            hipFuncAttributeMaxDynamicSharedMemorySize, (int)sh1);
  (void)hipFuncSetAttribute((const void*)pass2_kernel,
                            hipFuncAttributeMaxDynamicSharedMemorySize, (int)sh2);

  cvt_kernel<<<1024, 256, 0, stream>>>(s_in, sact, N_TOK * H_STU / 4);
  cvt_kernel<<<1024, 256, 0, stream>>>(t_in, tact, N_TOK * H_TEA / 4);
  pass1_kernel<<<NCHUNK, TPB, sh1, stream>>>(sact, tact, s_w, t_w,
                                             pm_s, pz_s, pm_t, pz_t);
  lse_kernel<<<(N_TOK + 255) / 256, 256, 0, stream>>>(pm_s, pz_s, lse_s);
  lse_kernel<<<(N_TOK + 255) / 256, 256, 0, stream>>>(pm_t, pz_t, lse_t);
  pass2_kernel<<<NCHUNK, TPB, sh2, stream>>>(sact, tact, s_w, t_w,
                                             lse_s, lse_t, lbl, prt_j, prt_h);
  finalize_kernel<<<1, 1, 0, stream>>>(prt_j, prt_h, out);
}
